// ScaledDotProductAttention_66005057405546
// MI455X (gfx1250) — matrix operand order repaired and checked by reading, not compiled, not run on hardware
//
#include <hip/hip_runtime.h>
#include <hip/hip_bf16.h>
#include <math.h>

// ---------------------------------------------------------------------------
// Distance-modulated attention for MI455X (gfx1250, wave32, WMMA).
// Per block: batch b, M=32 query rows (two 16-row WMMA tiles).
//   S  = relu(Q K^T)          [32 x 2048] bf16 in LDS
//   P  = (S @ R) / 8          R = (1+e)/(1+exp(1-dist)), pre-rescaled,
//                             pre-transposed bf16 (L2-resident, 8.4 MB)
//   out = softmax(P) @ V      flash-style online softmax per wave,
//                             cross-wave combine through LDS.
// Each Rt B-fragment load feeds 2 row-tiles (4 WMMAs / 128B of L2 traffic).
// ---------------------------------------------------------------------------

typedef __bf16 bf16_t;
typedef bf16_t v16bf __attribute__((ext_vector_type(16)));
typedef float  v8f   __attribute__((ext_vector_type(8)));

union FragB16 { v16bf v; int4 q[2]; };

#define NSEQ    2048
#define DHEAD   64
#define MROWS   32                      // query rows per block (2 WMMA tiles)
#define SSTRIDE 2056                    // 2048 + 8 bf16 pad (keeps 16B align)
#define PPSTR   40                      // expP bounce row stride (bf16)

static __device__ __forceinline__ v8f wmma_bf16(const FragB16& a, const FragB16& b, v8f c) {
    return __builtin_amdgcn_wmma_f32_16x16x32_bf16(false, a.v, false, b.v,
                                                   (short)0, c, false, false);
}

// ---------------- pre-pass: rescale + transpose dist -> Rt (bf16) -----------
__global__ void prep_rt_kernel(const float* __restrict__ dist, bf16_t* __restrict__ Rt) {
    __shared__ float tile[32][33];
    const int bx = blockIdx.x * 32;     // dist column (n)
    const int by = blockIdx.y * 32;     // dist row    (m)
    const int tx = threadIdx.x, ty = threadIdx.y;
    const float C = 1.0f + 2.718281828459045f;   // 1 + e
    #pragma unroll
    for (int i = 0; i < 32; i += 8) {
        float x = dist[(size_t)(by + ty + i) * NSEQ + bx + tx];
        tile[ty + i][tx] = C / (1.0f + __expf(1.0f - x));
    }
    __syncthreads();
    #pragma unroll
    for (int i = 0; i < 32; i += 8)     // Rt[n][m] = rescale(dist[m][n])
        Rt[(size_t)(bx + ty + i) * NSEQ + by + tx] = (bf16_t)tile[tx][ty + i];
}

// ---------------- pre-pass: f32 -> bf16 elementwise (Q, K) ------------------
__global__ void cvt_f32_bf16_kernel(const float* __restrict__ in, bf16_t* __restrict__ o, int n4) {
    int i = blockIdx.x * blockDim.x + threadIdx.x;
    if (i < n4) {
        float4 v = ((const float4*)in)[i];
        o[i * 4 + 0] = (bf16_t)v.x;
        o[i * 4 + 1] = (bf16_t)v.y;
        o[i * 4 + 2] = (bf16_t)v.z;
        o[i * 4 + 3] = (bf16_t)v.w;
    }
}

// ---------------- pre-pass: V [B,m,d] -> Vt [B,d,m] bf16 --------------------
__global__ void prep_vt_kernel(const float* __restrict__ V, bf16_t* __restrict__ Vt) {
    __shared__ float tile[32][33];
    const int b  = blockIdx.z;
    const int d0 = blockIdx.x * 32;     // 0 or 32
    const int m0 = blockIdx.y * 32;
    const int tx = threadIdx.x, ty = threadIdx.y;
    #pragma unroll
    for (int i = 0; i < 32; i += 8)
        tile[ty + i][tx] = V[((size_t)b * NSEQ + (m0 + ty + i)) * DHEAD + d0 + tx];
    __syncthreads();
    #pragma unroll
    for (int i = 0; i < 32; i += 8)
        Vt[((size_t)b * DHEAD + (d0 + ty + i)) * NSEQ + m0 + tx] = (bf16_t)tile[tx][ty + i];
}

// ---------------- main fused attention kernel -------------------------------
__global__ __launch_bounds__(256, 1)
void attn_main_kernel(const bf16_t* __restrict__ Qbf, const bf16_t* __restrict__ Kbf,
                      const bf16_t* __restrict__ Vt,  const bf16_t* __restrict__ Rt,
                      float* __restrict__ out) {
    __shared__ alignas(16) bf16_t sQ[MROWS * DHEAD];        //   4 KB
    __shared__ alignas(16) bf16_t sS[MROWS * SSTRIDE];      // 131.6 KB (reused for combine)
    __shared__ alignas(16) bf16_t sPP[8 * MROWS * PPSTR];   //  20 KB per-wave expP bounce

    const int tid  = threadIdx.x;
    const int lane = tid & 31;
    const int w    = tid >> 5;           // wave 0..7
    const int half = lane >> 4;          // 0: lanes 0-15, 1: lanes 16-31
    const int ln   = lane & 15;

    const int b  = blockIdx.x >> 6;              // batch (8 * 64 blocks)
    const int i0 = (blockIdx.x & 63) << 5;       // query-row base (32 rows)

    const bf16_t* Qb = Qbf + ((size_t)b * NSEQ + i0) * DHEAD;
    const bf16_t* Kb = Kbf + (size_t)b * NSEQ * DHEAD;
    const bf16_t* Vb = Vt  + (size_t)b * DHEAD * NSEQ;

    // ---- load Q tile (32x64 bf16 = 256 int4) into LDS ----
    ((int4*)sQ)[tid] = ((const int4*)Qb)[tid];
    __syncthreads();

    // ---- Phase 1: S = relu(Q K^T) -> bf16 in sS (two 16-row tiles) ----
    FragB16 aq[4];   // [rowtile0: k0,k1][rowtile1: k0,k1]
    {
        const bf16_t* q0 = sQ + (size_t)ln * DHEAD + half * 8;
        const bf16_t* q1 = sQ + (size_t)(16 + ln) * DHEAD + half * 8;
        aq[0].q[0] = *(const int4*)(q0 + 0);  aq[0].q[1] = *(const int4*)(q0 + 16);
        aq[1].q[0] = *(const int4*)(q0 + 32); aq[1].q[1] = *(const int4*)(q0 + 48);
        aq[2].q[0] = *(const int4*)(q1 + 0);  aq[2].q[1] = *(const int4*)(q1 + 16);
        aq[3].q[0] = *(const int4*)(q1 + 32); aq[3].q[1] = *(const int4*)(q1 + 48);
    }
    for (int jt = w; jt < 128; jt += 8) {
        const int j0 = jt << 4;
        const bf16_t* krow = Kb + (size_t)(j0 + ln) * DHEAD + half * 8;
        FragB16 b0, b1;
        b0.q[0] = *(const int4*)(krow + 0);
        b0.q[1] = *(const int4*)(krow + 16);
        b1.q[0] = *(const int4*)(krow + 32);
        b1.q[1] = *(const int4*)(krow + 48);
        v8f acc0 = {}, acc1 = {};
        acc0 = wmma_bf16(aq[0], b0, acc0);
        acc1 = wmma_bf16(aq[2], b0, acc1);
        acc0 = wmma_bf16(aq[1], b1, acc0);
        acc1 = wmma_bf16(aq[3], b1, acc1);
        bf16_t* srow0 = sS + (size_t)(half * 8) * SSTRIDE + j0 + ln;
        bf16_t* srow1 = sS + (size_t)(16 + half * 8) * SSTRIDE + j0 + ln;
        #pragma unroll
        for (int r = 0; r < 8; ++r) {
            float v0 = acc0[r], v1 = acc1[r];
            srow0[(size_t)r * SSTRIDE] = (bf16_t)(v0 > 0.0f ? v0 : 0.0f);
            srow1[(size_t)r * SSTRIDE] = (bf16_t)(v1 > 0.0f ? v1 : 0.0f);
        }
    }
    __syncthreads();

    // ---- Phase 2: P = (S @ R)/8 with online softmax + attn@V ----
    v8f   oAcc[2][4] = {};               // [rowtile][d-tile] 32x64 accumulator
    float mrun[2][8], lrun[2][8];
    #pragma unroll
    for (int rt = 0; rt < 2; ++rt)
        #pragma unroll
        for (int r = 0; r < 8; ++r) { mrun[rt][r] = -1e30f; lrun[rt][r] = 0.0f; }

    bf16_t* myPP = sPP + (size_t)w * MROWS * PPSTR;

    #pragma unroll 1
    for (int s = 0; s < 8; ++s) {
        const int jstrip = (w + 8 * s) << 5;     // 32-wide column strip
        v8f acc[2][2] = {};                       // [rowtile][coltile]
        const bf16_t* r0base = Rt + (size_t)(jstrip + ln) * NSEQ + half * 8;
        const bf16_t* r1base = Rt + (size_t)(jstrip + 16 + ln) * NSEQ + half * 8;
        const bf16_t* arow0  = sS + (size_t)ln * SSTRIDE + half * 8;
        const bf16_t* arow1  = sS + (size_t)(16 + ln) * SSTRIDE + half * 8;

        #pragma unroll 2
        for (int kk = 0; kk < NSEQ; kk += 32) {
            FragB16 af0, af1, bf0, bf1;
            af0.q[0] = *(const int4*)(arow0 + kk);
            af0.q[1] = *(const int4*)(arow0 + kk + 16);
            af1.q[0] = *(const int4*)(arow1 + kk);
            af1.q[1] = *(const int4*)(arow1 + kk + 16);
            bf0.q[0] = *(const int4*)(r0base + kk);
            bf0.q[1] = *(const int4*)(r0base + kk + 16);
            bf1.q[0] = *(const int4*)(r1base + kk);
            bf1.q[1] = *(const int4*)(r1base + kk + 16);
            acc[0][0] = wmma_bf16(af0, bf0, acc[0][0]);
            acc[1][0] = wmma_bf16(af1, bf0, acc[1][0]);
            acc[0][1] = wmma_bf16(af0, bf1, acc[0][1]);
            acc[1][1] = wmma_bf16(af1, bf1, acc[1][1]);
        }

        // online softmax per row tile; write exp(P) strip to per-wave bounce
        #pragma unroll
        for (int rt = 0; rt < 2; ++rt) {
            float e0[8], e1[8], newm[8], rowsum[8];
            #pragma unroll
            for (int r = 0; r < 8; ++r) {
                float p0 = acc[rt][0][r] * 0.125f;
                float p1 = acc[rt][1][r] * 0.125f;
                float t = fmaxf(p0, p1);
                #pragma unroll
                for (int off = 1; off < 16; off <<= 1)
                    t = fmaxf(t, __shfl_xor(t, off, 32));
                float mn = fmaxf(mrun[rt][r], t);
                newm[r] = mn;
                e0[r] = __expf(p0 - mn);
                e1[r] = __expf(p1 - mn);
                float rs = e0[r] + e1[r];
                #pragma unroll
                for (int off = 1; off < 16; off <<= 1)
                    rs += __shfl_xor(rs, off, 32);
                rowsum[r] = rs;
            }
            #pragma unroll
            for (int r = 0; r < 8; ++r) {
                float alpha = __expf(mrun[rt][r] - newm[r]);
                lrun[rt][r] = lrun[rt][r] * alpha + rowsum[r];
                mrun[rt][r] = newm[r];
                oAcc[rt][0][r] *= alpha; oAcc[rt][1][r] *= alpha;
                oAcc[rt][2][r] *= alpha; oAcc[rt][3][r] *= alpha;
            }
            bf16_t* prow = myPP + (size_t)(rt * 16 + half * 8) * PPSTR + ln;
            #pragma unroll
            for (int r = 0; r < 8; ++r) {
                prow[(size_t)r * PPSTR]      = (bf16_t)e0[r];
                prow[(size_t)r * PPSTR + 16] = (bf16_t)e1[r];
            }
        }
        asm volatile("s_wait_dscnt 0" ::: "memory");

        FragB16 ep0, ep1;
        {
            const bf16_t* p0 = myPP + (size_t)ln * PPSTR + half * 8;
            const bf16_t* p1 = myPP + (size_t)(16 + ln) * PPSTR + half * 8;
            ep0.q[0] = *(const int4*)(p0);
            ep0.q[1] = *(const int4*)(p0 + 16);
            ep1.q[0] = *(const int4*)(p1);
            ep1.q[1] = *(const int4*)(p1 + 16);
        }
        // O += exp(P) @ V_strip  (4 d-tiles; V frag shared by both row tiles)
        #pragma unroll
        for (int t = 0; t < 4; ++t) {
            const bf16_t* vrow = Vb + (size_t)(t * 16 + ln) * NSEQ + jstrip + half * 8;
            FragB16 bv;
            bv.q[0] = *(const int4*)(vrow);
            bv.q[1] = *(const int4*)(vrow + 16);
            oAcc[0][t] = wmma_bf16(ep0, bv, oAcc[0][t]);
            oAcc[1][t] = wmma_bf16(ep1, bv, oAcc[1][t]);
        }
    }

    // ---- Phase 3: cross-wave softmax combine (alias over dead sS) ----
    __syncthreads();
    float* cm = reinterpret_cast<float*>(sS);        // [8][32]
    float* cl = cm + 8 * MROWS;                      // [8][32]
    float* cO = cl + 8 * MROWS;                      // [8][32][64]
    #pragma unroll
    for (int rt = 0; rt < 2; ++rt) {
        float* orow = cO + ((size_t)w * MROWS + rt * 16 + half * 8) * DHEAD + ln;
        #pragma unroll
        for (int r = 0; r < 8; ++r) {
            orow[(size_t)r * DHEAD +  0] = oAcc[rt][0][r];
            orow[(size_t)r * DHEAD + 16] = oAcc[rt][1][r];
            orow[(size_t)r * DHEAD + 32] = oAcc[rt][2][r];
            orow[(size_t)r * DHEAD + 48] = oAcc[rt][3][r];
        }
        if (ln == 0) {
            #pragma unroll
            for (int r = 0; r < 8; ++r) {
                cm[w * MROWS + rt * 16 + half * 8 + r] = mrun[rt][r];
                cl[w * MROWS + rt * 16 + half * 8 + r] = lrun[rt][r];
            }
        }
    }
    __syncthreads();
    {
        const int row = tid >> 3;            // 0..31
        const int c0  = (tid & 7) * 8;       // 0..56
        float M = -1e30f;
        #pragma unroll
        for (int ww = 0; ww < 8; ++ww) M = fmaxf(M, cm[ww * MROWS + row]);
        float L = 0.0f;
        float v[8] = {};
        #pragma unroll
        for (int ww = 0; ww < 8; ++ww) {
            float sc = __expf(cm[ww * MROWS + row] - M);
            L += cl[ww * MROWS + row] * sc;
            const float* po = cO + ((size_t)ww * MROWS + row) * DHEAD + c0;
            #pragma unroll
            for (int c = 0; c < 8; ++c) v[c] += po[c] * sc;
        }
        const float invL = 1.0f / L;
        float* op = out + ((size_t)b * NSEQ + i0 + row) * DHEAD + c0;
        float4 r0 = make_float4(v[0] * invL, v[1] * invL, v[2] * invL, v[3] * invL);
        float4 r1 = make_float4(v[4] * invL, v[5] * invL, v[6] * invL, v[7] * invL);
        *(float4*)(op)     = r0;
        *(float4*)(op + 4) = r1;
    }
}

// ---------------------------------------------------------------------------
extern "C" void kernel_launch(void* const* d_in, const int* in_sizes, int n_in,
                              void* d_out, int out_size, void* d_ws, size_t ws_size,
                              hipStream_t stream) {
    const float* Q    = (const float*)d_in[0];
    const float* K    = (const float*)d_in[1];
    const float* V    = (const float*)d_in[2];
    const float* dist = (const float*)d_in[3];
    float* out = (float*)d_out;

    unsigned char* ws = (unsigned char*)d_ws;
    // workspace layout (bytes):
    bf16_t* Rt  = (bf16_t*)(ws);                                  // 2048*2048*2 = 8388608
    bf16_t* Qbf = (bf16_t*)(ws + 8388608);                        // 8*2048*64*2 = 2097152
    bf16_t* Kbf = (bf16_t*)(ws + 8388608 + 2097152);              // 2097152
    bf16_t* Vtb = (bf16_t*)(ws + 8388608 + 2 * 2097152);          // 2097152  (total ~14.6MB)

    // 1) rescale + transpose distance matrix -> bf16 (L2-resident afterwards)
    prep_rt_kernel<<<dim3(64, 64), dim3(32, 8), 0, stream>>>(dist, Rt);

    // 2) convert Q, K to bf16 (row-major), V to transposed bf16 [B,d,m]
    const int n4 = (8 * NSEQ * DHEAD) / 4;   // 262144
    cvt_f32_bf16_kernel<<<(n4 + 255) / 256, 256, 0, stream>>>(Q, Qbf, n4);
    cvt_f32_bf16_kernel<<<(n4 + 255) / 256, 256, 0, stream>>>(K, Kbf, n4);
    prep_vt_kernel<<<dim3(2, 64, 8), dim3(32, 8), 0, stream>>>(V, Vtb);

    // 3) fused attention: one block per (batch, 32 query rows)
    attn_main_kernel<<<8 * (NSEQ / MROWS), 256, 0, stream>>>(Qbf, Kbf, Vtb, Rt, out);
}
